// GCNEncoder_scale_degree_35201551958714
// MI455X (gfx1250) — matrix operand order repaired and checked by reading, not compiled, not run on hardware
//
#include <hip/hip_runtime.h>

#define N_NODES 100000
#define N_EDGES 1600000
#define IN_CH   256
#define OUT_CH  128
#define M_TILE  160      // 100000 = 625 * 160, exact tiling
#define M_SUB   (M_TILE / 16)

typedef __attribute__((ext_vector_type(16))) _Float16 v16h;
typedef __attribute__((ext_vector_type(8)))  _Float16 v8h;
typedef __attribute__((ext_vector_type(8)))  float    v8f;

// ---------------- degree / normalization prep ----------------

__global__ void gcn_init_deg(unsigned* __restrict__ deg) {
    int i = blockIdx.x * blockDim.x + threadIdx.x;
    if (i < N_NODES) deg[i] = 1u;   // self-loop
}

__global__ void gcn_count_deg(const int* __restrict__ col, unsigned* __restrict__ deg) {
    int e = blockIdx.x * blockDim.x + threadIdx.x;
    if (e < N_EDGES) atomicAdd(&deg[col[e]], 1u);
}

__global__ void gcn_dinv(const unsigned* __restrict__ deg, float* __restrict__ dinv) {
    int i = blockIdx.x * blockDim.x + threadIdx.x;
    if (i < N_NODES) dinv[i] = rsqrtf((float)deg[i]);   // deg >= 1 always
}

// ---- h = x @ W via WMMA f16->f32 (160-row M-tile), fused agg = b + h*dinv^2 ----

__global__ __launch_bounds__(256) void gcn_gemm_wmma(
    const float* __restrict__ x, const float* __restrict__ W,
    const float* __restrict__ b, const float* __restrict__ dinv,
    float* __restrict__ h, float* __restrict__ agg)
{
    extern __shared__ _Float16 smem[];
    _Float16* ldsA  = smem;                  // [160][256] x tile (m,k) f16, 80 KB
    _Float16* ldsBT = smem + M_TILE * 256;   // [128][256] W transposed (n,k) f16, 64 KB

    const int tid = threadIdx.x;
    const int m0  = blockIdx.x * M_TILE;

    // stage x tile: 160x256 f32 -> f16 (vectorized, coalesced); 40 float4 per thread
    const float4* xt = (const float4*)(x + (size_t)m0 * IN_CH);
    #pragma unroll 8
    for (int i = 0; i < (M_TILE * IN_CH / 4) / 256; ++i) {
        int idx4 = tid + i * 256;
        float4 v = xt[idx4];
        int o = idx4 * 4;
        ldsA[o + 0] = (_Float16)v.x; ldsA[o + 1] = (_Float16)v.y;
        ldsA[o + 2] = (_Float16)v.z; ldsA[o + 3] = (_Float16)v.w;
    }
    // stage W transposed: W[k][n] (row-major) -> ldsBT[n][k] f16; 128 per thread
    #pragma unroll 8
    for (int i = 0; i < 128; ++i) {
        int id = tid + i * 256;              // 32768 elements
        int k = id >> 7, n = id & 127;
        ldsBT[n * 256 + k] = (_Float16)W[id];
    }
    __syncthreads();

    const int wave  = tid >> 5;              // n-tile 0..7
    const int lane  = tid & 31;
    const int group = lane >> 4;             // 0 or 1
    const int mloc  = lane & 15;
    const int ncol  = wave * 16 + mloc;      // this lane's output column

    // hoist all 8 B fragments (K = kc + group*16 + e, contiguous in W^T)
    v16h bf[8];
    #pragma unroll
    for (int kk = 0; kk < 8; ++kk)
        bf[kk] = *(const v16h*)(ldsBT + ncol * 256 + kk * 32 + group * 16);

    const float bias = b[ncol];

    for (int ms = 0; ms < M_SUB; ++ms) {     // 10 M-subtiles, B frags reused
        const int mbase = ms * 16;
        v8f acc = {};
        #pragma unroll
        for (int kk = 0; kk < 8; ++kk) {
            // A frag: halves 0..15 at K = kc+group*16+e (contiguous, matches B)
            const _Float16* ap = ldsA + (mbase + mloc) * 256 + kk * 32 + group * 16;
            v8h a0 = *(const v8h*)(ap);
            v8h a1 = *(const v8h*)(ap + 8);
            v16h a = __builtin_shufflevector(a0, a1,
                     0,1,2,3,4,5,6,7,8,9,10,11,12,13,14,15);
            acc = __builtin_amdgcn_wmma_f32_16x16x32_f16(
                      false, a, false, bf[kk], (short)0, acc, false, false);
        }
        // C/D layout: VGPR r, lanes 0-15 -> M=r, lanes 16-31 -> M=8+r; N = lane&15
        #pragma unroll
        for (int r = 0; r < 8; ++r) {
            int m = m0 + mbase + r + group * 8;
            float hv = acc[r];
            float di = dinv[m];
            h  [(size_t)m * OUT_CH + ncol] = hv;
            agg[(size_t)m * OUT_CH + ncol] = bias + hv * di * di;  // bias + self-loop
        }
    }
}

// ---------------- edge scatter: agg[col] += h[row] * dinv[row]*dinv[col] ----

#define PF_DIST 64

__global__ void gcn_edge_scatter(const int* __restrict__ row, const int* __restrict__ col,
                                 const float* __restrict__ h, const float* __restrict__ dinv,
                                 float* __restrict__ agg)
{
    int t = blockIdx.x * blockDim.x + threadIdx.x;
    int e = t >> 5;                      // one wave32 per edge
    int lane = threadIdx.x & 31;
    if (e >= N_EDGES) return;

    // warm L2/L0 for an upcoming edge's gathered row (global_prefetch_b8)
    if (e + PF_DIST < N_EDGES) {
        int rn = row[e + PF_DIST];
        __builtin_prefetch(h + (size_t)rn * OUT_CH + lane * 4, 0, 0);
    }

    int r = row[e], c = col[e];
    float w = dinv[r] * dinv[c];
    float4 hv = ((const float4*)(h + (size_t)r * OUT_CH))[lane];  // coalesced 512B/wave
    float* dst = agg + (size_t)c * OUT_CH + lane * 4;
    atomicAdd(dst + 0, hv.x * w);
    atomicAdd(dst + 1, hv.y * w);
    atomicAdd(dst + 2, hv.z * w);
    atomicAdd(dst + 3, hv.w * w);
}

// ---------------- per-node min-max scale + L2 normalize ----------------

__global__ void gcn_finalize(const float* __restrict__ agg, float* __restrict__ out)
{
    int t = blockIdx.x * blockDim.x + threadIdx.x;
    int node = t >> 5;                   // one wave32 per node, 4 ch per lane
    int lane = threadIdx.x & 31;
    if (node >= N_NODES) return;
    float4 v = ((const float4*)(agg + (size_t)node * OUT_CH))[lane];

    float mn = fminf(fminf(v.x, v.y), fminf(v.z, v.w));
    float mx = fmaxf(fmaxf(v.x, v.y), fmaxf(v.z, v.w));
    #pragma unroll
    for (int off = 16; off > 0; off >>= 1) {
        mn = fminf(mn, __shfl_xor(mn, off, 32));
        mx = fmaxf(mx, __shfl_xor(mx, off, 32));
    }
    float inv = 1.0f / (mx - mn);
    float4 z;
    z.x = (v.x - mn) * inv; z.y = (v.y - mn) * inv;
    z.z = (v.z - mn) * inv; z.w = (v.w - mn) * inv;

    float ss = z.x * z.x + z.y * z.y + z.z * z.z + z.w * z.w;
    #pragma unroll
    for (int off = 16; off > 0; off >>= 1) ss += __shfl_xor(ss, off, 32);
    float s = 1.0f / fmaxf(sqrtf(ss), 1e-12f);

    float4 o; o.x = z.x * s; o.y = z.y * s; o.z = z.z * s; o.w = z.w * s;
    ((float4*)(out + (size_t)node * OUT_CH))[lane] = o;
}

// ---------------- launch ----------------

extern "C" void kernel_launch(void* const* d_in, const int* in_sizes, int n_in,
                              void* d_out, int out_size, void* d_ws, size_t ws_size,
                              hipStream_t stream)
{
    const float* x  = (const float*)d_in[0];
    const int*   ei = (const int*)d_in[1];      // [2, N_EDGES]: row then col
    const float* W  = (const float*)d_in[2];
    const float* b  = (const float*)d_in[3];
    float* out = (float*)d_out;

    char* ws = (char*)d_ws;
    float*    h    = (float*)ws;                                          // 51.2 MB
    float*    agg  = (float*)(ws + (size_t)N_NODES * OUT_CH * 4);         // 51.2 MB
    unsigned* deg  = (unsigned*)(ws + 2 * (size_t)N_NODES * OUT_CH * 4);  // 400 KB
    float*    dinv = (float*)(ws + 2 * (size_t)N_NODES * OUT_CH * 4
                                 + (size_t)N_NODES * 4);                  // 400 KB

    const int* row = ei;
    const int* col = ei + N_EDGES;

    gcn_init_deg <<<(N_NODES + 255) / 256, 256, 0, stream>>>(deg);
    gcn_count_deg<<<(N_EDGES + 255) / 256, 256, 0, stream>>>(col, deg);
    gcn_dinv     <<<(N_NODES + 255) / 256, 256, 0, stream>>>(deg, dinv);

    // 625 M-tiles of 160 rows; 144 KB dynamic LDS (x tile f16 + W^T f16)
    gcn_gemm_wmma<<<N_NODES / M_TILE, 256,
                    (M_TILE * 256 + 128 * 256) * 2, stream>>>(x, W, b, dinv, h, agg);

    // one wave per edge: 1.6M waves
    gcn_edge_scatter<<<(N_EDGES * 32 + 255) / 256, 256, 0, stream>>>(
        row, col, h, dinv, agg);

    // one wave per node
    gcn_finalize<<<(N_NODES * 32 + 255) / 256, 256, 0, stream>>>(agg, out);
}
